// GCNIDS_60344290509379
// MI455X (gfx1250) — compile-verified, matrix operand order checked by reading, not run on hardware
//
#include <hip/hip_runtime.h>
#include <hip/hip_bf16.h>

// ---------------------------------------------------------------------------
// GCN forward on MI455X (gfx1250): bf16 WMMA GEMMs (32x64 register-blocked) +
// L2-resident edge scatter aggregation + fused BN/ReLU.
// ---------------------------------------------------------------------------

typedef __bf16 bf16_t;
typedef __attribute__((ext_vector_type(16))) __bf16 v16bf;
typedef __attribute__((ext_vector_type(8)))  float  v8f;

#define GCN_D_IN   128
#define GCN_D_HID  256
#define GCN_D_OUT  128
#define GCN_N_CLS  10
#define GCN_EPS    1e-5f

static inline int cdiv(int a, int b) { return (a + b - 1) / b; }

// --------------------------- utility kernels -------------------------------

__global__ void k_fill(float* __restrict__ p, float v, int n) {
    int i = blockIdx.x * blockDim.x + threadIdx.x;
    if (i < n) p[i] = v;
}

__global__ void k_deg_accum(const long long* __restrict__ dst,
                            float* __restrict__ deg, int E) {
    int i = blockIdx.x * blockDim.x + threadIdx.x;
    if (i < E) atomicAdd(&deg[(int)dst[i]], 1.0f);
}

__global__ void k_to_rsqrt(float* __restrict__ p, int n) {
    int i = blockIdx.x * blockDim.x + threadIdx.x;
    if (i < n) p[i] = rsqrtf(p[i]);   // deg >= 1 (self loop) always
}

// ------------------------ WMMA operand packing -----------------------------
// Packed tile layout: [tile][kt][lane][16 bf16] ; 512 bf16 (1KB) per 16x32 tile.
// Per-lane element e -> K = kt*32 + (e<8 ? e : e+8) + 8*(lane>=16); row/col = lane&15.

__global__ void k_pack_a(const float* __restrict__ A, bf16_t* __restrict__ P,
                         int M, int K) {
    int tid = blockIdx.x * blockDim.x + threadIdx.x;
    int ktn = K >> 5;
    int total = (M >> 4) * ktn * 32;
    if (tid >= total) return;
    int lane = tid & 31;
    int t    = tid >> 5;
    int kt   = t % ktn;
    int mt   = t / ktn;
    int m  = mt * 16 + (lane & 15);
    int k0 = kt * 32 + ((lane >> 4) << 3);
    const float* src = A + (size_t)m * K + k0;
    float4 lo0 = *(const float4*)(src);
    float4 lo1 = *(const float4*)(src + 4);
    float4 hi0 = *(const float4*)(src + 16);
    float4 hi1 = *(const float4*)(src + 20);
    v16bf o;
    o[0]  = (bf16_t)lo0.x; o[1]  = (bf16_t)lo0.y; o[2]  = (bf16_t)lo0.z; o[3]  = (bf16_t)lo0.w;
    o[4]  = (bf16_t)lo1.x; o[5]  = (bf16_t)lo1.y; o[6]  = (bf16_t)lo1.z; o[7]  = (bf16_t)lo1.w;
    o[8]  = (bf16_t)hi0.x; o[9]  = (bf16_t)hi0.y; o[10] = (bf16_t)hi0.z; o[11] = (bf16_t)hi0.w;
    o[12] = (bf16_t)hi1.x; o[13] = (bf16_t)hi1.y; o[14] = (bf16_t)hi1.z; o[15] = (bf16_t)hi1.w;
    *(v16bf*)(P + (size_t)tid * 16) = o;
}

__global__ void k_pack_w(const float* __restrict__ W, bf16_t* __restrict__ P,
                         int K, int Nout) {
    int tid = blockIdx.x * blockDim.x + threadIdx.x;
    int ktn = K >> 5;
    int total = (Nout >> 4) * ktn * 32;
    if (tid >= total) return;
    int lane = tid & 31;
    int t    = tid >> 5;
    int kt   = t % ktn;
    int nt   = t / ktn;
    int n  = nt * 16 + (lane & 15);
    int k0 = kt * 32 + ((lane >> 4) << 3);
    v16bf o;
#pragma unroll
    for (int e = 0; e < 8; ++e)
        o[e] = (bf16_t)W[(size_t)(k0 + e) * Nout + n];
#pragma unroll
    for (int e = 0; e < 8; ++e)
        o[8 + e] = (bf16_t)W[(size_t)(k0 + 16 + e) * Nout + n];
    *(v16bf*)(P + (size_t)tid * 16) = o;
}

// ----------------------------- WMMA GEMM -----------------------------------
// One wave computes a 32(M) x 64(N) strip: 2 A fragments x 4 B fragments ->
// 8 WMMAs per K step (128 KFLOP per 6KB of L2 fragments, 1.5 loads/WMMA).
// Early-out is wave-uniform, so active waves keep EXEC all-ones for WMMA.

__global__ void k_gemm_wmma(const bf16_t* __restrict__ Apk,
                            const bf16_t* __restrict__ Wpk,
                            float* __restrict__ C, int M, int K, int Nout) {
    int wave   = (blockIdx.x * blockDim.x + threadIdx.x) >> 5;
    int lane   = threadIdx.x & 31;
    int ntn    = Nout >> 4;
    int nquads = ntn >> 2;               // N tiles processed 4 at a time
    int mt     = wave / nquads;          // index of 32-row M block
    int nq     = wave % nquads;
    if (mt >= (M >> 5)) return;          // uniform across the wave
    int ktn = K >> 5;

    size_t aTile = (size_t)ktn * 512;    // bf16 elems per packed 16-row strip
    const bf16_t* aBase0 = Apk + (size_t)(2 * mt) * aTile + (size_t)lane * 16;
    const bf16_t* aBase1 = aBase0 + aTile;
    const bf16_t* bBase  = Wpk + (size_t)(nq * 4) * aTile + (size_t)lane * 16;

    v8f a0c0 = {}, a0c1 = {}, a0c2 = {}, a0c3 = {};
    v8f a1c0 = {}, a1c1 = {}, a1c2 = {}, a1c3 = {};
    for (int kt = 0; kt < ktn; ++kt) {
        size_t ko = (size_t)kt * 512;
        v16bf a0 = *(const v16bf*)(aBase0 + ko);
        v16bf a1 = *(const v16bf*)(aBase1 + ko);
        v16bf b0 = *(const v16bf*)(bBase + ko);
        v16bf b1 = *(const v16bf*)(bBase + aTile     + ko);
        v16bf b2 = *(const v16bf*)(bBase + aTile * 2 + ko);
        v16bf b3 = *(const v16bf*)(bBase + aTile * 3 + ko);
        a0c0 = __builtin_amdgcn_wmma_f32_16x16x32_bf16(false, a0, false, b0, (short)0, a0c0, false, false);
        a1c0 = __builtin_amdgcn_wmma_f32_16x16x32_bf16(false, a1, false, b0, (short)0, a1c0, false, false);
        a0c1 = __builtin_amdgcn_wmma_f32_16x16x32_bf16(false, a0, false, b1, (short)0, a0c1, false, false);
        a1c1 = __builtin_amdgcn_wmma_f32_16x16x32_bf16(false, a1, false, b1, (short)0, a1c1, false, false);
        a0c2 = __builtin_amdgcn_wmma_f32_16x16x32_bf16(false, a0, false, b2, (short)0, a0c2, false, false);
        a1c2 = __builtin_amdgcn_wmma_f32_16x16x32_bf16(false, a1, false, b2, (short)0, a1c2, false, false);
        a0c3 = __builtin_amdgcn_wmma_f32_16x16x32_bf16(false, a0, false, b3, (short)0, a0c3, false, false);
        a1c3 = __builtin_amdgcn_wmma_f32_16x16x32_bf16(false, a1, false, b3, (short)0, a1c3, false, false);
    }

    // C layout per tile: VGPR i -> row tile*16 + i + 8*(lane>=16), col nt*16 + (lane&15)
    int mbase = mt * 32 + ((lane >> 4) << 3);
    int ncol  = (nq * 4) * 16 + (lane & 15);
    float* c0 = C + (size_t)mbase * Nout + ncol;
    float* c1 = c0 + (size_t)16 * Nout;
#pragma unroll
    for (int i = 0; i < 8; ++i) {
        float* cr0 = c0 + (size_t)i * Nout;
        float* cr1 = c1 + (size_t)i * Nout;
        cr0[0]  = a0c0[i];
        cr0[16] = a0c1[i];
        cr0[32] = a0c2[i];
        cr0[48] = a0c3[i];
        cr1[0]  = a1c0[i];
        cr1[16] = a1c1[i];
        cr1[32] = a1c2[i];
        cr1[48] = a1c3[i];
    }
}

// --------------------------- graph aggregation -----------------------------

// out[i,f] = h[i,f] * dinv[i]^2 + bias[f]   (self loop + bias)
__global__ void k_agg_init(const float* __restrict__ h,
                           const float* __restrict__ dinv,
                           const float* __restrict__ bias,
                           float* __restrict__ out, int D, int N) {
    int tid = blockIdx.x * blockDim.x + threadIdx.x;
    if (tid >= N * D) return;
    int nrow = tid / D;
    int f    = tid - nrow * D;
    float di = dinv[nrow];
    out[tid] = h[tid] * di * di + bias[f];
}

// out[dst,f] += h[src,f] * dinv[src]*dinv[dst]  (L2-resident atomics)
__global__ void k_agg_edges(const float* __restrict__ h,
                            const long long* __restrict__ src,
                            const long long* __restrict__ dst,
                            const float* __restrict__ dinv,
                            float* __restrict__ out, int D, int E) {
    int tid    = blockIdx.x * blockDim.x + threadIdx.x;
    int chunks = D >> 2;                 // float4 chunks per edge
    if (tid >= E * chunks) return;
    int e = tid / chunks;
    int c = tid - e * chunks;
    int s = (int)src[e];
    int d = (int)dst[e];
    float nrm = dinv[s] * dinv[d];
    float4 hv = *(const float4*)(h + (size_t)s * D + c * 4);
    float* o  = out + (size_t)d * D + c * 4;
    atomicAdd(o + 0, hv.x * nrm);
    atomicAdd(o + 1, hv.y * nrm);
    atomicAdd(o + 2, hv.z * nrm);
    atomicAdd(o + 3, hv.w * nrm);
}

// ----------------------------- batch norm ----------------------------------

// blockDim.x == D; each block reduces a row-chunk, coalesced across columns.
__global__ void k_bn_stats(const float* __restrict__ x,
                           float* __restrict__ sum, float* __restrict__ sumsq,
                           int D, int N, int rowsPerBlock) {
    int f  = threadIdx.x;
    int r0 = blockIdx.x * rowsPerBlock;
    int r1 = r0 + rowsPerBlock; if (r1 > N) r1 = N;
    float s = 0.f, ss = 0.f;
    for (int r = r0; r < r1; ++r) {
        float v = x[(size_t)r * D + f];
        s += v; ss += v * v;
    }
    atomicAdd(&sum[f], s);
    atomicAdd(&sumsq[f], ss);
}

__global__ void k_bn_relu(const float* __restrict__ x,
                          const float* __restrict__ sum,
                          const float* __restrict__ sumsq,
                          const float* __restrict__ gamma,
                          const float* __restrict__ beta,
                          float* __restrict__ y, int D, int N) {
    int tid = blockIdx.x * blockDim.x + threadIdx.x;
    if (tid >= N * D) return;
    int f = tid % D;
    float invN = 1.0f / (float)N;
    float mu   = sum[f] * invN;
    float var  = sumsq[f] * invN - mu * mu;
    float v = gamma[f] * (x[tid] - mu) * rsqrtf(var + GCN_EPS) + beta[f];
    y[tid] = v > 0.f ? v : 0.f;
}

// --------------------------- output projection -----------------------------

__global__ void k_out_linear(const float* __restrict__ h,
                             const float* __restrict__ Wout,
                             const float* __restrict__ bout,
                             float* __restrict__ out, int N) {
    int tid = blockIdx.x * blockDim.x + threadIdx.x;
    if (tid >= N * GCN_N_CLS) return;
    int n = tid / GCN_N_CLS;
    int c = tid - n * GCN_N_CLS;
    const float* hr = h + (size_t)n * GCN_D_OUT;
    float acc = bout[c];
#pragma unroll 4
    for (int k = 0; k < GCN_D_OUT; k += 4) {
        float4 hv = *(const float4*)(hr + k);
        acc += hv.x * Wout[(k + 0) * GCN_N_CLS + c];
        acc += hv.y * Wout[(k + 1) * GCN_N_CLS + c];
        acc += hv.z * Wout[(k + 2) * GCN_N_CLS + c];
        acc += hv.w * Wout[(k + 3) * GCN_N_CLS + c];
    }
    out[tid] = acc;
}

// ------------------------------ host driver --------------------------------

static void run_layer(const float* in, int K, int Dout,
                      const float* W, const float* bias,
                      const float* gamma, const float* beta,
                      const long long* src, const long long* dst,
                      const float* dinv,
                      float* gemm_out, float* agg_out, float* act_out,
                      bf16_t* packA, bf16_t* packW, float* stats,
                      int N, int E, hipStream_t stream) {
    int ktn = K / 32, ntn = Dout / 16, mtn = N / 16;

    int packA_threads = mtn * ktn * 32;
    k_pack_a<<<cdiv(packA_threads, 256), 256, 0, stream>>>(in, packA, N, K);

    int packW_threads = ntn * ktn * 32;
    k_pack_w<<<cdiv(packW_threads, 256), 256, 0, stream>>>(W, packW, K, Dout);

    int gemm_threads = (N / 32) * (ntn / 4) * 32;   // one wave per 32x64 C strip
    k_gemm_wmma<<<cdiv(gemm_threads, 256), 256, 0, stream>>>(
        packA, packW, gemm_out, N, K, Dout);

    k_agg_init<<<cdiv(N * Dout, 256), 256, 0, stream>>>(
        gemm_out, dinv, bias, agg_out, Dout, N);

    k_agg_edges<<<cdiv(E * (Dout / 4), 256), 256, 0, stream>>>(
        gemm_out, src, dst, dinv, agg_out, Dout, E);

    k_fill<<<cdiv(2 * Dout, 256), 256, 0, stream>>>(stats, 0.f, 2 * Dout);
    k_bn_stats<<<cdiv(N, 256), Dout, 0, stream>>>(
        agg_out, stats, stats + Dout, Dout, N, 256);
    k_bn_relu<<<cdiv(N * Dout, 256), 256, 0, stream>>>(
        agg_out, stats, stats + Dout, gamma, beta, act_out, Dout, N);
}

extern "C" void kernel_launch(void* const* d_in, const int* in_sizes, int n_in,
                              void* d_out, int out_size, void* d_ws, size_t ws_size,
                              hipStream_t stream) {
    const float*     x    = (const float*)d_in[0];
    const long long* ei   = (const long long*)d_in[1];
    const float*     W1   = (const float*)d_in[2];
    const float*     b1   = (const float*)d_in[3];
    const float*     g1   = (const float*)d_in[4];
    const float*     be1  = (const float*)d_in[5];
    const float*     W2   = (const float*)d_in[6];
    const float*     b2   = (const float*)d_in[7];
    const float*     g2   = (const float*)d_in[8];
    const float*     be2  = (const float*)d_in[9];
    const float*     W3   = (const float*)d_in[10];
    const float*     b3   = (const float*)d_in[11];
    const float*     g3   = (const float*)d_in[12];
    const float*     be3  = (const float*)d_in[13];
    const float*     Wout = (const float*)d_in[14];
    const float*     bout = (const float*)d_in[15];
    float*           out  = (float*)d_out;

    const int N = in_sizes[0] / GCN_D_IN;     // 100000
    const int E = in_sizes[1] / 2;            // 1600000
    const long long* src = ei;
    const long long* dst = ei + E;

    // workspace carve-up (256B aligned)
    char* ws = (char*)d_ws;
    size_t off = 0;
    auto take = [&](size_t bytes) {
        char* p = ws + off;
        off += (bytes + 255) & ~(size_t)255;
        return p;
    };
    float*  dinv  = (float*)take((size_t)N * 4);
    float*  bufA  = (float*)take((size_t)N * GCN_D_HID * 4);
    float*  bufB  = (float*)take((size_t)N * GCN_D_HID * 4);
    bf16_t* packA = (bf16_t*)take((size_t)N * GCN_D_HID * 2);
    bf16_t* packW = (bf16_t*)take((size_t)GCN_D_HID * GCN_D_HID * 2);
    float*  stats = (float*)take((size_t)2 * GCN_D_HID * 4);
    (void)ws_size;

    // symmetric-normalization degrees (with self loops): dinv = rsqrt(indeg + 1)
    k_fill<<<cdiv(N, 256), 256, 0, stream>>>(dinv, 1.0f, N);
    k_deg_accum<<<cdiv(E, 256), 256, 0, stream>>>(dst, dinv, E);
    k_to_rsqrt<<<cdiv(N, 256), 256, 0, stream>>>(dinv, N);

    // layer 1: x(N,128) -> bufA act (N,256)
    run_layer(x, GCN_D_IN, GCN_D_HID, W1, b1, g1, be1, src, dst, dinv,
              /*gemm*/bufA, /*agg*/bufB, /*act*/bufA,
              packA, packW, stats, N, E, stream);

    // layer 2: bufA(N,256) -> bufB act (N,256)
    run_layer(bufA, GCN_D_HID, GCN_D_HID, W2, b2, g2, be2, src, dst, dinv,
              /*gemm*/bufB, /*agg*/bufA, /*act*/bufB,
              packA, packW, stats, N, E, stream);

    // layer 3: bufB(N,256) -> bufA act (N,128)
    run_layer(bufB, GCN_D_HID, GCN_D_OUT, W3, b3, g3, be3, src, dst, dinv,
              /*gemm*/bufA, /*agg*/bufB, /*act*/bufA,
              packA, packW, stats, N, E, stream);

    // output projection: (N,128) @ (128,10) + bias
    k_out_linear<<<cdiv(N * GCN_N_CLS, 256), 256, 0, stream>>>(
        bufA, Wout, bout, out, N);
}
